// OneGNN_10797547782337
// MI455X (gfx1250) — matrix operand order repaired and checked by reading, not compiled, not run on hardware
//
#include <hip/hip_runtime.h>
#include <hip/hip_bf16.h>

// ---------------------------------------------------------------------------
// OneGNN for MI455X (gfx1250, wave32).
//  - Node GEMMs: v_wmma_f32_16x16x32_bf16, weights staged in LDS via
//    GLOBAL_LOAD_ASYNC_TO_LDS_B128 (ASYNCcnt) with plain-copy fallback.
//  - Edge scatter / graph pool: HW f32 global atomics (L2-resident buffers).
// All launches on `stream`; no malloc/sync (graph-capture safe).
// ---------------------------------------------------------------------------

typedef __bf16 bf16;
typedef __attribute__((ext_vector_type(16))) __bf16 v16bf;
typedef __attribute__((ext_vector_type(8)))  float  v8f;
typedef __attribute__((ext_vector_type(4)))  int    v4i;

#define NNODES  50000
#define NEDGES  800000
#define DIM     96
#define NLAYERS 3
#define NGRAPHS 128
#define OUTDIM  10

// gfx1250 async global->LDS copy path (guarded; falls back to plain copy)
#ifdef __has_builtin
#if __has_builtin(__builtin_amdgcn_global_load_async_to_lds_b128) && \
    __has_builtin(__builtin_amdgcn_s_wait_asynccnt)
#define USE_ASYNC_LDS 1
#endif
#endif
#ifndef USE_ASYNC_LDS
#define USE_ASYNC_LDS 0
#endif

// ---------------- WMMA fragment helpers (ISA 7.12.2, wave32) ----------------

// A: 16x32 bf16 (MxK). Lane l (l<16): row M=l, K = k0+{0..7,16..23};
// lane l+16: row M=l, K = k0+{8..15,24..31}.
__device__ __forceinline__ v16bf load_a_frag(const bf16* __restrict__ A,
                                             int row0, int k0, int lane) {
  const int m  = lane & 15;
  const int hi = lane >> 4;
  const bf16* p = A + (size_t)(row0 + m) * DIM;
  const int kA = k0 + 8 * hi;
  const int kB = k0 + 16 + 8 * hi;
  v16bf a;
#pragma unroll
  for (int j = 0; j < 8; ++j) {
    a[j]     = p[kA + j];
    a[8 + j] = p[kB + j];
  }
  return a;
}

// B: 32x16 bf16 (KxN). Lane l: N = l&15; lanes 0-15 K=k0+{0..7,16..23}, lanes 16-31
// K=k0+{8..15,24..31}. Wt layout is transposed: Wt[n*DIM + k]. Works for global or LDS.
__device__ __forceinline__ v16bf load_b_frag(const bf16* __restrict__ Wt,
                                             int n0, int k0, int lane) {
  const int n  = lane & 15;
  const int hi = lane >> 4;
  const bf16* p = Wt + (size_t)(n0 + n) * DIM + k0 + 8 * hi;
  v16bf b;
#pragma unroll
  for (int j = 0; j < 8; ++j) {
    b[j]     = p[j];
    b[8 + j] = p[16 + j];
  }
  return b;
}

__device__ __forceinline__ v8f wmma_bf16(v16bf a, v16bf b, v8f c) {
  // (neg_a, A, neg_b, B, c_mod, C, reuse_a, reuse_b)
  return __builtin_amdgcn_wmma_f32_16x16x32_bf16(false, a, false, b,
                                                 (short)0, c, false, false);
}

// C/D: 16x16 f32. Lane l: N = l&15; VGPR r holds row M = 8*(l>>4)+r.
__device__ __forceinline__ void store_c(float* __restrict__ out, int row0,
                                        int n0, int lane, v8f c) {
  const int n  = lane & 15;
  const int hi = lane >> 4;
#pragma unroll
  for (int r = 0; r < 8; ++r)
    out[(size_t)(row0 + 8 * hi + r) * DIM + n0 + n] = c[r];
}

// Stage one 96x96 bf16 matrix (18432 B = 1152 x 16B) into LDS.
__device__ __forceinline__ void stage_w(const bf16* __restrict__ g,
                                        bf16* __restrict__ s) {
  const int nvec = DIM * DIM * (int)sizeof(bf16) / 16;  // 1152
#if USE_ASYNC_LDS
  typedef __attribute__((address_space(1))) v4i* gp_t;
  typedef __attribute__((address_space(3))) v4i* lp_t;
  gp_t gp = (gp_t)g;
  lp_t lp = (lp_t)s;
  for (int i = threadIdx.x; i < nvec; i += blockDim.x)
    __builtin_amdgcn_global_load_async_to_lds_b128(gp + i, lp + i, 0, 0);
#else
  const uint4* gv = (const uint4*)g;
  uint4* sv = (uint4*)s;
  for (int i = threadIdx.x; i < nvec; i += blockDim.x) sv[i] = gv[i];
#endif
}

__device__ __forceinline__ void stage_fence() {
#if USE_ASYNC_LDS
  __builtin_amdgcn_s_wait_asynccnt(0);
#endif
  __syncthreads();
}

// --------------------------------- kernels ---------------------------------

__global__ void k_cvt_bf16(const float* __restrict__ in, bf16* __restrict__ out,
                           int n) {
  int tid = blockIdx.x * blockDim.x + threadIdx.x;
  if (tid < n) out[tid] = (bf16)in[tid];
}

// Wt[n*DIM+k] = (bf16) W[k*DIM+n]
__global__ void k_transpose_w(const float* __restrict__ W,
                              bf16* __restrict__ Wt) {
  int tid = blockIdx.x * blockDim.x + threadIdx.x;
  if (tid >= DIM * DIM) return;
  int n = tid / DIM, k = tid - n * DIM;
  Wt[tid] = (bf16)W[(size_t)k * DIM + n];
}

// One wave per 16-row tile: agg = h@W1, msg = h@W2. Weights in LDS.
__global__ __launch_bounds__(256) void k_gemm_dual(
    const bf16* __restrict__ hb, const bf16* __restrict__ Wt1,
    const bf16* __restrict__ Wt2, float* __restrict__ agg,
    float* __restrict__ msg, int nrows) {
  __shared__ bf16 sW[2 * DIM * DIM];  // 36864 B of 320KB/WGP

  stage_w(Wt1, sW);
  stage_w(Wt2, sW + DIM * DIM);
  stage_fence();

  const int lane = threadIdx.x & 31;
  const int tile = blockIdx.x * (blockDim.x >> 5) + (threadIdx.x >> 5);
  const int row0 = tile * 16;
  if (row0 >= nrows) return;  // wave-uniform, after barrier: EXEC all-ones

  v16bf a0 = load_a_frag(hb, row0, 0,  lane);
  v16bf a1 = load_a_frag(hb, row0, 32, lane);
  v16bf a2 = load_a_frag(hb, row0, 64, lane);

#pragma unroll
  for (int w = 0; w < 2; ++w) {
    const bf16* Wt = sW + w * DIM * DIM;
    float* out     = w ? msg : agg;
#pragma unroll
    for (int nt = 0; nt < 6; ++nt) {
      v8f acc = {0.f, 0.f, 0.f, 0.f, 0.f, 0.f, 0.f, 0.f};
      acc = wmma_bf16(a0, load_b_frag(Wt, nt * 16, 0,  lane), acc);
      acc = wmma_bf16(a1, load_b_frag(Wt, nt * 16, 32, lane), acc);
      acc = wmma_bf16(a2, load_b_frag(Wt, nt * 16, 64, lane), acc);
      store_c(out, row0, nt * 16, lane, acc);
    }
  }
}

// agg[dst[e]][d] += msg[src[e]][d]  (HW f32 atomics; msg/agg are L2-resident)
__global__ void k_scatter(const int* __restrict__ src,
                          const int* __restrict__ dst,
                          const float* __restrict__ msg,
                          float* __restrict__ agg, int nedges) {
  int tid = blockIdx.x * blockDim.x + threadIdx.x;
  if (tid >= nedges * DIM) return;
  int e = tid / DIM;
  int d = tid - e * DIM;
  float v = msg[(size_t)src[e] * DIM + d];
  unsafeAtomicAdd(agg + (size_t)dst[e] * DIM + d, v);
}

// hb = bf16(relu(agg))
__global__ void k_relu_cvt(const float* __restrict__ agg,
                           bf16* __restrict__ hb, int n) {
  int tid = blockIdx.x * blockDim.x + threadIdx.x;
  if (tid >= n) return;
  float v = agg[tid];
  hb[tid] = (bf16)(v > 0.f ? v : 0.f);
}

__global__ void k_zero(float* __restrict__ p, int n) {
  int tid = blockIdx.x * blockDim.x + threadIdx.x;
  if (tid < n) p[tid] = 0.f;
}

// gemb[batch[node]][d] += relu(agg[node][d])
__global__ void k_pool(const float* __restrict__ agg,
                       const int* __restrict__ batch,
                       float* __restrict__ gemb, int n) {
  int tid = blockIdx.x * blockDim.x + threadIdx.x;
  if (tid >= n) return;
  int node = tid / DIM;
  int d    = tid - node * DIM;
  float v = agg[tid];
  v = v > 0.f ? v : 0.f;
  unsafeAtomicAdd(gemb + (size_t)batch[node] * DIM + d, v);
}

// hidden = relu(gemb @ cw1 + cb1): 8 row-tiles x 6 col-tiles = 48 WMMA waves
__global__ __launch_bounds__(128) void k_gemm_cls(
    const bf16* __restrict__ Ab, const bf16* __restrict__ Wt,
    const float* __restrict__ bias, float* __restrict__ out, int nrows) {
  const int lane = threadIdx.x & 31;
  const int wg   = blockIdx.x * (blockDim.x >> 5) + (threadIdx.x >> 5);
  const int nt   = wg % 6;
  const int rt   = wg / 6;
  const int row0 = rt * 16;
  if (row0 >= nrows) return;

  v16bf a0 = load_a_frag(Ab, row0, 0,  lane);
  v16bf a1 = load_a_frag(Ab, row0, 32, lane);
  v16bf a2 = load_a_frag(Ab, row0, 64, lane);

  v8f acc = {0.f, 0.f, 0.f, 0.f, 0.f, 0.f, 0.f, 0.f};
  acc = wmma_bf16(a0, load_b_frag(Wt, nt * 16, 0,  lane), acc);
  acc = wmma_bf16(a1, load_b_frag(Wt, nt * 16, 32, lane), acc);
  acc = wmma_bf16(a2, load_b_frag(Wt, nt * 16, 64, lane), acc);

  const int n  = lane & 15;
  const int hi = lane >> 4;
  const float b = bias[nt * 16 + n];
#pragma unroll
  for (int r = 0; r < 8; ++r) {
    float v = acc[r] + b;
    out[(size_t)(row0 + 8 * hi + r) * DIM + nt * 16 + n] = v > 0.f ? v : 0.f;
  }
}

// out[g][o] = hidden[g] . cw2[:,o] + cb2[o]
__global__ void k_cls2(const float* __restrict__ hidden,
                       const float* __restrict__ cw2,
                       const float* __restrict__ cb2,
                       float* __restrict__ out) {
  int tid = blockIdx.x * blockDim.x + threadIdx.x;
  if (tid >= NGRAPHS * OUTDIM) return;
  int g = tid / OUTDIM;
  int o = tid - g * OUTDIM;
  float s = cb2[o];
#pragma unroll 8
  for (int k = 0; k < DIM; ++k)
    s += hidden[(size_t)g * DIM + k] * cw2[k * OUTDIM + o];
  out[tid] = s;
}

// ------------------------------- host driver -------------------------------

extern "C" void kernel_launch(void* const* d_in, const int* in_sizes, int n_in,
                              void* d_out, int out_size, void* d_ws, size_t ws_size,
                              hipStream_t stream) {
  const float* x     = (const float*)d_in[0];
  const int*   edge  = (const int*)d_in[1];
  const int*   src   = edge;            // edge_index[0]
  const int*   dst   = edge + NEDGES;   // edge_index[1]
  const int*   batch = (const int*)d_in[2];
  const float* W1    = (const float*)d_in[3];  // [3,96,96]
  const float* W2    = (const float*)d_in[4];  // [3,96,96]
  const float* cw1   = (const float*)d_in[5];
  const float* cb1   = (const float*)d_in[6];
  const float* cw2   = (const float*)d_in[7];
  const float* cb2   = (const float*)d_in[8];
  float* out = (float*)d_out;

  char* ws = (char*)d_ws;
  auto carve = [&](size_t bytes) {
    char* p = ws;
    ws += (bytes + 255) & ~(size_t)255;
    return p;
  };
  bf16*  hb     = (bf16*) carve((size_t)NNODES * DIM * sizeof(bf16));
  float* agg    = (float*)carve((size_t)NNODES * DIM * sizeof(float));
  float* msg    = (float*)carve((size_t)NNODES * DIM * sizeof(float));
  bf16*  Wt1    = (bf16*) carve((size_t)DIM * DIM * sizeof(bf16));
  bf16*  Wt2    = (bf16*) carve((size_t)DIM * DIM * sizeof(bf16));
  bf16*  cw1t   = (bf16*) carve((size_t)DIM * DIM * sizeof(bf16));
  float* gemb   = (float*)carve((size_t)NGRAPHS * DIM * sizeof(float));
  bf16*  gembb  = (bf16*) carve((size_t)NGRAPHS * DIM * sizeof(bf16));
  float* hidden = (float*)carve((size_t)NGRAPHS * DIM * sizeof(float));

  const int nfeat = NNODES * DIM;  // 4.8M
  const int TB = 256;

  // h0 = bf16(x)
  k_cvt_bf16<<<(nfeat + TB - 1) / TB, TB, 0, stream>>>(x, hb, nfeat);

  const int ntiles = NNODES / 16;            // 3125 (exact)
  const int gemm_blocks = (ntiles + 7) / 8;  // 8 waves / block share LDS weights

  for (int l = 0; l < NLAYERS; ++l) {
    const float* W1l = W1 + (size_t)l * DIM * DIM;
    const float* W2l = W2 + (size_t)l * DIM * DIM;
    k_transpose_w<<<(DIM * DIM + TB - 1) / TB, TB, 0, stream>>>(W1l, Wt1);
    k_transpose_w<<<(DIM * DIM + TB - 1) / TB, TB, 0, stream>>>(W2l, Wt2);

    // agg = h@W1 (self part), msg = h@W2
    k_gemm_dual<<<gemm_blocks, 256, 0, stream>>>(hb, Wt1, Wt2, agg, msg, NNODES);

    // agg += scatter(msg[src] -> dst)
    const long long sthreads = (long long)NEDGES * DIM;  // 76.8M
    k_scatter<<<(int)((sthreads + TB - 1) / TB), TB, 0, stream>>>(src, dst, msg,
                                                                 agg, NEDGES);

    if (l < NLAYERS - 1)
      k_relu_cvt<<<(nfeat + TB - 1) / TB, TB, 0, stream>>>(agg, hb, nfeat);
  }

  // graph pooling (relu fused)
  const int gfeat = NGRAPHS * DIM;  // 12288
  k_zero<<<(gfeat + TB - 1) / TB, TB, 0, stream>>>(gemb, gfeat);
  k_pool<<<(nfeat + TB - 1) / TB, TB, 0, stream>>>(agg, batch, gemb, nfeat);

  // hidden = relu(gemb @ cw1 + cb1) via WMMA
  k_cvt_bf16<<<(gfeat + TB - 1) / TB, TB, 0, stream>>>(gemb, gembb, gfeat);
  k_transpose_w<<<(DIM * DIM + TB - 1) / TB, TB, 0, stream>>>(cw1, cw1t);
  k_gemm_cls<<<12, 128, 0, stream>>>(gembb, cw1t, cb1, hidden, NGRAPHS);

  // out = hidden @ cw2 + cb2
  k_cls2<<<(NGRAPHS * OUTDIM + TB - 1) / TB, TB, 0, stream>>>(hidden, cw2, cb2,
                                                              out);
  (void)in_sizes; (void)n_in; (void)out_size; (void)ws_size;
}
